// Transformer_23776938951494
// MI455X (gfx1250) — matrix operand order repaired and checked by reading, not compiled, not run on hardware
//
#include <hip/hip_runtime.h>
#include <cstddef>

// ---------------- CDNA5 (gfx1250) WMMA types ----------------
typedef __attribute__((ext_vector_type(16))) __bf16 v16bf;
typedef __attribute__((ext_vector_type(8)))  __bf16 v8bf;
typedef __attribute__((ext_vector_type(4)))  __bf16 v4bf;
typedef __attribute__((ext_vector_type(8)))  float  v8f;

#define DEV __device__ __forceinline__

static constexpr int Bsz   = 2;
static constexpr int Nseq  = 1024;
static constexpr int Dm    = 1024;
static constexpr int Hh    = 16;
static constexpr int DH    = 64;
static constexpr int INNER = Hh * DH;     // 1024
static constexpr int FF    = 4096;
static constexpr int Lnum  = 6;
static constexpr int Mrows = Bsz * Nseq;  // 2048

DEV v8f wmma_bf16(v16bf a, v16bf b, v8f c) {
  // D = A(16x32 bf16) * B(32x16 bf16) + C(16x16 f32)
  return __builtin_amdgcn_wmma_f32_16x16x32_bf16(
      /*neg_a=*/false, a, /*neg_b=*/false, b,
      /*c_mod=*/(short)0, c, /*reuse_a=*/false, /*reuse_b=*/false);
}

// A-matrix fragment (16x32 bf16): lane = row (t&15); K = 16g + {0..15} contiguous
// p must point at A[row*stride + k0 + 16*(t>>4)]
DEV v16bf frag_a_bf16(const __bf16* p) {
  v8bf lo = *reinterpret_cast<const v8bf*>(p);
  v8bf hi = *reinterpret_cast<const v8bf*>(p + 8);
  v16bf f;
#pragma unroll
  for (int i = 0; i < 8; ++i) { f[i] = lo[i]; f[i + 8] = hi[i]; }
  return f;
}

// B-matrix fragment (32x16 bf16): lane = col (t&15); K = 16*(t>>4) + {0..15} contiguous.
// p must point at Bcol[k0 + 16*(t>>4)] (single aligned 32B load)
DEV v16bf frag_b_bf16(const __bf16* p) {
  return *reinterpret_cast<const v16bf*>(p);
}

// ---------------- fp32 -> bf16 bulk conversion (per-layer weight staging) --------
__global__ __launch_bounds__(256) void cvt_bf16_kernel(const float* __restrict__ in,
                                                       __bf16* __restrict__ out) {
  const size_t i = ((size_t)blockIdx.x * 256 + threadIdx.x) * 4;
  float4 xv = *reinterpret_cast<const float4*>(in + i);
  v4bf o;
  o[0] = (__bf16)xv.x; o[1] = (__bf16)xv.y; o[2] = (__bf16)xv.z; o[3] = (__bf16)xv.w;
  *reinterpret_cast<v4bf*>(out + i) = o;
}

// ---------------- LayerNorm (fp32 in -> bf16 out) ----------------
__global__ __launch_bounds__(256) void ln_bf16_kernel(
    const float* __restrict__ x, const float* __restrict__ w,
    const float* __restrict__ b, __bf16* __restrict__ out) {
  const int row = blockIdx.x;                 // Mrows
  const int t   = threadIdx.x;                // 256, 4 elems each
  const float* xr = x + (size_t)row * Dm;
  float4 xv = reinterpret_cast<const float4*>(xr)[t];
  __shared__ float red[8];

  float s = xv.x + xv.y + xv.z + xv.w;
#pragma unroll
  for (int off = 16; off; off >>= 1) s += __shfl_xor(s, off, 32);
  if ((t & 31) == 0) red[t >> 5] = s;
  __syncthreads();
  s = 0.f;
#pragma unroll
  for (int i = 0; i < 8; ++i) s += red[i];
  const float mean = s * (1.f / (float)Dm);

  float dx = xv.x - mean, dy = xv.y - mean, dz = xv.z - mean, dw = xv.w - mean;
  float vs = dx * dx + dy * dy + dz * dz + dw * dw;
  __syncthreads();
#pragma unroll
  for (int off = 16; off; off >>= 1) vs += __shfl_xor(vs, off, 32);
  if ((t & 31) == 0) red[t >> 5] = vs;
  __syncthreads();
  vs = 0.f;
#pragma unroll
  for (int i = 0; i < 8; ++i) vs += red[i];
  const float inv = rsqrtf(vs * (1.f / (float)Dm) + 1e-5f);

  float4 wv = reinterpret_cast<const float4*>(w)[t];
  float4 bv = reinterpret_cast<const float4*>(b)[t];
  __bf16* op = out + (size_t)row * Dm + t * 4;
  op[0] = (__bf16)(dx * inv * wv.x + bv.x);
  op[1] = (__bf16)(dy * inv * wv.y + bv.y);
  op[2] = (__bf16)(dz * inv * wv.z + bv.z);
  op[3] = (__bf16)(dw * inv * wv.w + bv.w);
}

// ---------------- WMMA GEMM: C[M,O] = A[M,K](bf16) @ W[O,K](bf16)^T ----------
// flags: 1 = add bias[o], 2 = exact-erf GELU, 4 = add res[m*O+o]
// one wave computes a 32x64 output tile: 2 A-fragments reuse 4 shared B-fragments
__global__ __launch_bounds__(32) void wmma_gemm_kernel(
    const __bf16* __restrict__ A, const __bf16* __restrict__ W,
    const float* __restrict__ bias, const float* __restrict__ res,
    float* __restrict__ outF, __bf16* __restrict__ outB,
    int M, int K, int O, int flags) {
  const int t    = threadIdx.x;
  const int lane = t & 15;
  const int g    = t >> 4;
  const int kbA  = g * 16;
  const int kbB  = g * 16;
  const int m0   = blockIdx.x * 32;
  const int o0   = blockIdx.y * 64;

  v8f acc[2][4] = {};
  const __bf16* Ap0 = A + (size_t)(m0 + lane) * K + kbA;
  const __bf16* Ap1 = Ap0 + (size_t)16 * K;
  const __bf16* Wp[4];
#pragma unroll
  for (int j = 0; j < 4; ++j)
    Wp[j] = W + (size_t)(o0 + j * 16 + lane) * K + kbB;

  for (int k0 = 0; k0 < K; k0 += 32) {
    __builtin_prefetch(Wp[0] + k0 + 256, 0, 0);   // global_prefetch_b8 on weight stream
    v16bf a0 = frag_a_bf16(Ap0 + k0);
    v16bf a1 = frag_a_bf16(Ap1 + k0);
#pragma unroll
    for (int j = 0; j < 4; ++j) {
      v16bf bb = frag_b_bf16(Wp[j] + k0);
      acc[0][j] = wmma_bf16(a0, bb, acc[0][j]);
      acc[1][j] = wmma_bf16(a1, bb, acc[1][j]);
    }
  }

  // C/D layout: lane holds col (t&15), rows 8g+p
#pragma unroll
  for (int j = 0; j < 4; ++j) {
    const int o = o0 + j * 16 + lane;
    const float bval = (flags & 1) ? bias[o] : 0.f;
#pragma unroll
    for (int half = 0; half < 2; ++half) {
#pragma unroll
      for (int p = 0; p < 8; ++p) {
        const int m = m0 + half * 16 + g * 8 + p;
        float v = acc[half][j][p] + bval;
        if (flags & 2) v = 0.5f * v * (1.f + erff(v * 0.70710678118654752f));
        const size_t idx = (size_t)m * O + o;
        if (flags & 4) v += res[idx];
        if (outF) outF[idx] = v;
        if (outB) outB[idx] = (__bf16)v;
      }
    }
  }
}

// ---------------- Fused attention per (batch, head, 16-row q tile) ----------------
// q/k/v heads are contiguous [N,64] blocks (reference uses raw .reshape, no transpose).
__global__ __launch_bounds__(32) void attn_kernel(
    const __bf16* __restrict__ q, const __bf16* __restrict__ k,
    const __bf16* __restrict__ v, __bf16* __restrict__ o) {
  __shared__ float sc[16 * Nseq];            // 64 KB: 16 score rows
  const int t    = threadIdx.x;
  const int lane = t & 15;
  const int g    = t >> 4;
  const int kbA  = g * 16;
  const int kbB  = g * 16;
  const int tile = blockIdx.x;               // N/16
  const size_t hb = ((size_t)blockIdx.z * Hh + blockIdx.y) * (size_t)(Nseq * DH);
  const __bf16* Q = q + hb;
  const __bf16* K = k + hb;
  const __bf16* V = v + hb;

  // Q A-fragments over d: chunks d=[0,32) and [32,64)
  const __bf16* qrow = Q + (size_t)(tile * 16 + lane) * DH;
  const v16bf aq0 = frag_a_bf16(qrow + kbA);
  const v16bf aq1 = frag_a_bf16(qrow + 32 + kbA);

  // scores = (Q K^T) * 1/sqrt(DH), 64 key tiles of 16
  for (int j = 0; j < 64; ++j) {
    v8f acc = {};
    const __bf16* krow = K + (size_t)(j * 16 + lane) * DH;   // B col = key index
    acc = wmma_bf16(aq0, frag_b_bf16(krow + kbB), acc);
    acc = wmma_bf16(aq1, frag_b_bf16(krow + 32 + kbB), acc);
#pragma unroll
    for (int p = 0; p < 8; ++p)
      sc[(g * 8 + p) * Nseq + j * 16 + lane] = acc[p] * 0.125f;
  }
  __syncthreads();

  // softmax over each of the 16 rows (wave-wide shfl reductions)
  for (int row = 0; row < 16; ++row) {
    float* s = sc + row * Nseq;
    float mx = -3.4e38f;
    for (int i = t; i < Nseq; i += 32) mx = fmaxf(mx, s[i]);
#pragma unroll
    for (int off = 16; off; off >>= 1) mx = fmaxf(mx, __shfl_xor(mx, off, 32));
    float sum = 0.f;
    for (int i = t; i < Nseq; i += 32) { float e = __expf(s[i] - mx); s[i] = e; sum += e; }
#pragma unroll
    for (int off = 16; off; off >>= 1) sum += __shfl_xor(sum, off, 32);
    const float inv = 1.f / sum;
    for (int i = t; i < Nseq; i += 32) s[i] *= inv;
  }
  __syncthreads();

  // O(16x64) = P(16x1024) @ V(1024x64)
  v8f oacc[4] = {};
  for (int k0 = 0; k0 < Nseq; k0 += 32) {
    const float* pr = sc + lane * Nseq + k0 + kbA;
    v16bf a;
#pragma unroll
    for (int i = 0; i < 8; ++i) { a[i] = (__bf16)pr[i]; a[i + 8] = (__bf16)pr[8 + i]; }
#pragma unroll
    for (int jd = 0; jd < 4; ++jd) {
      const __bf16* vp = V + (size_t)(k0 + kbB) * DH + jd * 16 + lane;  // B col = d
      v16bf bb;
#pragma unroll
      for (int i = 0; i < 16; ++i) bb[i] = vp[(size_t)i * DH];
      oacc[jd] = wmma_bf16(a, bb, oacc[jd]);
    }
  }
  __bf16* orow = o + hb + (size_t)tile * 16 * DH;
#pragma unroll
  for (int jd = 0; jd < 4; ++jd)
#pragma unroll
    for (int p = 0; p < 8; ++p)
      orow[(size_t)(g * 8 + p) * DH + jd * 16 + lane] = (__bf16)oacc[jd][p];
}

// ---------------- final mean over sequence ----------------
__global__ __launch_bounds__(256) void mean_kernel(const float* __restrict__ x,
                                                   float* __restrict__ out) {
  const int idx = blockIdx.x * blockDim.x + threadIdx.x;   // B*D = 2048
  const int b = idx / Dm, d = idx % Dm;
  const float* p = x + (size_t)b * Nseq * Dm + d;
  float s = 0.f;
  for (int n = 0; n < Nseq; ++n) s += p[(size_t)n * Dm];
  out[idx] = s * (1.f / (float)Nseq);
}

// ---------------- host driver ----------------
extern "C" void kernel_launch(void* const* d_in, const int* in_sizes, int n_in,
                              void* d_out, int out_size, void* d_ws, size_t ws_size,
                              hipStream_t stream) {
  (void)in_sizes; (void)n_in; (void)out_size; (void)ws_size;
  const float* x_in = (const float*)d_in[0];
  const float* ln1w = (const float*)d_in[1];
  const float* ln1b = (const float*)d_in[2];
  const float* WQ   = (const float*)d_in[3];
  const float* WK   = (const float*)d_in[4];
  const float* WV   = (const float*)d_in[5];
  const float* WO   = (const float*)d_in[6];
  const float* ln2w = (const float*)d_in[7];
  const float* ln2b = (const float*)d_in[8];
  const float* W1   = (const float*)d_in[9];
  const float* b1   = (const float*)d_in[10];
  const float* W2   = (const float*)d_in[11];
  const float* b2   = (const float*)d_in[12];

  char* ws = (char*)d_ws;
  size_t off = 0;
  float*  xbuf = (float*)(ws + off);  off += (size_t)Mrows * Dm * sizeof(float);   // 8 MB
  __bf16* hbuf = (__bf16*)(ws + off); off += (size_t)Mrows * Dm * 2;               // 4 MB
  __bf16* qbuf = (__bf16*)(ws + off); off += (size_t)Mrows * INNER * 2;            // 4 MB
  __bf16* kbuf = (__bf16*)(ws + off); off += (size_t)Mrows * INNER * 2;            // 4 MB
  __bf16* vbuf = (__bf16*)(ws + off); off += (size_t)Mrows * INNER * 2;            // 4 MB
  __bf16* obuf = (__bf16*)(ws + off); off += (size_t)Mrows * INNER * 2;            // 4 MB
  __bf16* fbuf = (__bf16*)(ws + off); off += (size_t)Mrows * FF * 2;               // 16 MB
  __bf16* wscr = (__bf16*)(ws + off); off += (size_t)FF * Dm * 2;                  //  8 MB weight staging

  hipMemcpyAsync(xbuf, x_in, (size_t)Mrows * Dm * sizeof(float),
                 hipMemcpyDeviceToDevice, stream);

  const dim3 gSquare(Mrows / 32, INNER / 64);   // (64,16) for 2048x1024 outputs
  const dim3 gFF(Mrows / 32, FF / 64);          // (64,64) for 2048x4096
  const dim3 gAttn(Nseq / 16, Hh, Bsz);         // (64,16,2)
  const int cvtSq = (Dm * INNER) / (256 * 4);   // 1024 blocks for 1M-elem weights
  const int cvtFF = (FF * Dm) / (256 * 4);      // 4096 blocks for 4M-elem weights

  for (int i = 0; i < Lnum; ++i) {
    const size_t wsq = (size_t)i * INNER * Dm;
    // LN1 -> bf16
    ln_bf16_kernel<<<Mrows, 256, 0, stream>>>(xbuf, ln1w + i * Dm, ln1b + i * Dm, hbuf);
    // Q, K, V projections (bf16 outputs); stage each weight as bf16 first
    cvt_bf16_kernel<<<cvtSq, 256, 0, stream>>>(WQ + wsq, wscr);
    wmma_gemm_kernel<<<gSquare, 32, 0, stream>>>(hbuf, wscr, nullptr, nullptr,
                                                 nullptr, qbuf, Mrows, Dm, INNER, 0);
    cvt_bf16_kernel<<<cvtSq, 256, 0, stream>>>(WK + wsq, wscr);
    wmma_gemm_kernel<<<gSquare, 32, 0, stream>>>(hbuf, wscr, nullptr, nullptr,
                                                 nullptr, kbuf, Mrows, Dm, INNER, 0);
    cvt_bf16_kernel<<<cvtSq, 256, 0, stream>>>(WV + wsq, wscr);
    wmma_gemm_kernel<<<gSquare, 32, 0, stream>>>(hbuf, wscr, nullptr, nullptr,
                                                 nullptr, vbuf, Mrows, Dm, INNER, 0);
    // fused softmax attention
    attn_kernel<<<gAttn, 32, 0, stream>>>(qbuf, kbuf, vbuf, obuf);
    // output projection + residual (fp32, in place)
    cvt_bf16_kernel<<<cvtSq, 256, 0, stream>>>(WO + (size_t)i * Dm * INNER, wscr);
    wmma_gemm_kernel<<<gSquare, 32, 0, stream>>>(obuf, wscr, nullptr, xbuf,
                                                 xbuf, nullptr, Mrows, INNER, Dm, 4);
    // LN2 -> bf16
    ln_bf16_kernel<<<Mrows, 256, 0, stream>>>(xbuf, ln2w + i * Dm, ln2b + i * Dm, hbuf);
    // FFN up + bias + GELU (bf16 out)
    cvt_bf16_kernel<<<cvtFF, 256, 0, stream>>>(W1 + (size_t)i * FF * Dm, wscr);
    wmma_gemm_kernel<<<gFF, 32, 0, stream>>>(hbuf, wscr, b1 + (size_t)i * FF, nullptr,
                                             nullptr, fbuf, Mrows, Dm, FF, 1 | 2);
    // FFN down + bias + residual (fp32, in place)
    cvt_bf16_kernel<<<cvtFF, 256, 0, stream>>>(W2 + (size_t)i * Dm * FF, wscr);
    wmma_gemm_kernel<<<gSquare, 32, 0, stream>>>(fbuf, wscr, b2 + (size_t)i * Dm, xbuf,
                                                 xbuf, nullptr, Mrows, FF, Dm, 1 | 4);
  }
  mean_kernel<<<(Bsz * Dm) / 256, 256, 0, stream>>>(xbuf, (float*)d_out);
}
